// Attention_83820581748737
// MI455X (gfx1250) — compile-verified, matrix operand order checked by reading, not run on hardware
//
#include <hip/hip_runtime.h>
#include <stdint.h>

#define DDIM 256
#define SLEN 2048
#define BATCH 2
#define NHEAD 8

typedef __attribute__((ext_vector_type(16))) __bf16 v16bf;
typedef __attribute__((ext_vector_type(8)))  float  v8f;

__device__ __forceinline__ __bf16 bits2bf(unsigned short u) {
  return __builtin_bit_cast(__bf16, u);
}
__device__ __forceinline__ unsigned short f2bfbits(float f) {
  unsigned int u = __builtin_bit_cast(unsigned int, f);
  u += 0x7FFFu + ((u >> 16) & 1u);   // round-to-nearest-even
  return (unsigned short)(u >> 16);
}

// ----- CDNA5 async global->LDS copy (CDNA5 ISA, GV mode) -----------
__device__ __forceinline__ uint32_t lds_addr32(const void* p) {
  return (uint32_t)(uintptr_t)p;  // generic LDS pointer: low 32 bits = LDS offset
}
__device__ __forceinline__ void async_ld_b128(uint32_t lds_off, uint64_t gaddr) {
  asm volatile("global_load_async_to_lds_b128 %0, %1, off"
               :: "v"(lds_off), "v"(gaddr) : "memory");
}
// async loads complete in order -> wait ASYNCcnt<=16 retires the older group of 16
__device__ __forceinline__ void wait_async_all() {
  asm volatile("s_wait_asynccnt 0" ::: "memory");
}
__device__ __forceinline__ void wait_async_prev16() {
  asm volatile("s_wait_asynccnt 16" ::: "memory");
}

// A-matrix fragment (16x32 bf16). Element (m,k) at base[m*ld + k] (k contiguous).
__device__ __forceinline__ v16bf frag_a(const unsigned short* base, int ld, int lane) {
  const int m  = lane & 15;
  const int ko = (lane >> 4) << 3;          // 0 or 8
  v16bf a;
#pragma unroll
  for (int r = 0; r < 8; ++r) {
    const int k = ((r & 4) << 2) + ko + ((r & 3) << 1);
    a[2 * r]     = bits2bf(base[m * ld + k]);
    a[2 * r + 1] = bits2bf(base[m * ld + k + 1]);
  }
  return a;
}

// B-matrix fragment (32x16 bf16). Element (k,n) at base[n*ld + k] (k contiguous).
__device__ __forceinline__ v16bf frag_b(const unsigned short* base, int ld, int lane) {
  const int n  = lane & 15;
  const int ko = (lane >> 4) << 3;
  v16bf b;
#pragma unroll
  for (int r = 0; r < 8; ++r) {
    const int k = ((r & 4) << 2) + ko + ((r & 3) << 1);
    b[2 * r]     = bits2bf(base[n * ld + k]);
    b[2 * r + 1] = bits2bf(base[n * ld + k + 1]);
  }
  return b;
}

// ---------------------------------------------------------------------------
// Projection: dst = src @ W (4096x256 @ 256x256), bf16 WMMA, f32 accumulate,
// plus f32 row norms for Q and K.  V stored transposed ([B][D][S]).
// grid = (32, 3), block = 256 (8 waves); blockIdx.y selects Q/K/V.
// ---------------------------------------------------------------------------
__global__ __launch_bounds__(256)
void proj_kernel(const float* __restrict__ p, const float* __restrict__ e,
                 const float* __restrict__ Wq, const float* __restrict__ Wk,
                 const float* __restrict__ Wv,
                 unsigned short* __restrict__ Qh, unsigned short* __restrict__ Kh,
                 unsigned short* __restrict__ Vt,
                 float* __restrict__ q2, float* __restrict__ k2) {
  const int mat = blockIdx.y;
  const float* src = (mat == 2) ? e : p;
  const float* W   = (mat == 0) ? Wq : (mat == 1) ? Wk : Wv;
  unsigned short* dst = (mat == 0) ? Qh : (mat == 1) ? Kh : Vt;
  float* nrm = (mat == 0) ? q2 : k2;

  const int row0 = blockIdx.x * 128;
  const int tid  = threadIdx.x;
  const int wave = tid >> 5;
  const int lane = tid & 31;

  __shared__ __align__(16) unsigned short As[128][32];   //  8 KB  (A: k contiguous)
  __shared__ __align__(16) unsigned short Bt[256][40];   // 20 KB  (W^T: k contiguous)

  v8f zero = {};
  v8f acc[16];
#pragma unroll
  for (int t = 0; t < 16; ++t) acc[t] = zero;

  for (int kp = 0; kp < 8; ++kp) {          // 8 k-panels of 32
    __syncthreads();
    // A panel: 128x32 f32 -> bf16, float4 loads + packed 64-bit LDS stores
    for (int i = tid; i < 1024; i += 256) {
      const int r = i >> 3, c4 = (i & 7) * 4;
      const float4 v = *reinterpret_cast<const float4*>(
          &src[(size_t)(row0 + r) * DDIM + kp * 32 + c4]);
      const uint64_t packed =
          (uint64_t)f2bfbits(v.x) | ((uint64_t)f2bfbits(v.y) << 16) |
          ((uint64_t)f2bfbits(v.z) << 32) | ((uint64_t)f2bfbits(v.w) << 48);
      *reinterpret_cast<uint64_t*>(&As[r][c4]) = packed;
    }
    // B panel: 32x256 f32 loaded row-major, stored transposed (Bt[n][k])
    for (int i = tid; i < 2048; i += 256) {
      const int k = i >> 6, n4 = (i & 63) * 4;
      const float4 v = *reinterpret_cast<const float4*>(
          &W[(size_t)(kp * 32 + k) * DDIM + n4]);
      Bt[n4 + 0][k] = f2bfbits(v.x);
      Bt[n4 + 1][k] = f2bfbits(v.y);
      Bt[n4 + 2][k] = f2bfbits(v.z);
      Bt[n4 + 3][k] = f2bfbits(v.w);
    }
    __syncthreads();
    v16bf a = frag_a(&As[wave * 16][0], 32, lane);
    // two interleaved B-fragment buffers: both live -> distinct registers,
    // each loads under the other's WMMA (partial dscnt waits)
    v16bf b0 = frag_b(&Bt[0][0], 40, lane);
    v16bf b1 = frag_b(&Bt[16][0], 40, lane);
#pragma unroll
    for (int t = 0; t < 16; t += 2) {
      acc[t] = __builtin_amdgcn_wmma_f32_16x16x32_bf16(
          false, a, false, b0, (short)0, acc[t], false, false);
      if (t + 2 < 16) b0 = frag_b(&Bt[(t + 2) * 16][0], 40, lane);
      acc[t + 1] = __builtin_amdgcn_wmma_f32_16x16x32_bf16(
          false, a, false, b1, (short)0, acc[t + 1], false, false);
      if (t + 3 < 16) b1 = frag_b(&Bt[(t + 3) * 16][0], 40, lane);
    }
  }

  const int n  = lane & 15;
  const int hi = lane >> 4;
#pragma unroll
  for (int r = 0; r < 8; ++r) {
    const int row = row0 + wave * 16 + r + 8 * hi;     // flat row = b*SLEN + s
    float sq = 0.0f;
#pragma unroll
    for (int t = 0; t < 16; ++t) {
      const float v = acc[t][r];
      const int col = t * 16 + n;
      if (mat < 2) {
        dst[(size_t)row * DDIM + col] = f2bfbits(v);
      } else {  // V transposed: [b][d][s]
        const int bb = row >> 11, ss = row & (SLEN - 1);
        dst[((size_t)bb * DDIM + col) * SLEN + ss] = f2bfbits(v);
      }
      sq += v * v;
    }
    sq += __shfl_xor(sq, 1, 32);
    sq += __shfl_xor(sq, 2, 32);
    sq += __shfl_xor(sq, 4, 32);
    sq += __shfl_xor(sq, 8, 32);
    if (mat < 2 && n == 0) nrm[row] = sq;
  }
}

// ---------------------------------------------------------------------------
// Fused RBF attention with double-buffered async K/V staging.
// grid = (64, 8, 2), block = 128 (4 waves); wave w: rows 16*(w>>1), D-cols 128*(w&1).
// ---------------------------------------------------------------------------
__global__ __launch_bounds__(128)
void attn_kernel(const unsigned short* __restrict__ Qh,
                 const unsigned short* __restrict__ Kh,
                 const unsigned short* __restrict__ Vt,
                 const float* __restrict__ q2, const float* __restrict__ k2,
                 const float* __restrict__ gamma, float* __restrict__ out) {
  const int rt = blockIdx.x;
  const int h  = blockIdx.y;
  const int b  = blockIdx.z;
  const int tid   = threadIdx.x;
  const int wave  = tid >> 5;
  const int lane  = tid & 31;
  const int rhalf = wave >> 1;    // row half (16 rows)
  const int chalf = wave & 1;     // gram key quadrant / D-col half

  __shared__ __align__(16) unsigned short Qs[32][256];      // 16 KB, [s][d]
  __shared__ __align__(16) unsigned short Ks[2][32][256];   // 32 KB, [t][d] x2
  __shared__ __align__(16) unsigned short VsT[2][256][40];  // 40 KB, [d][t] x2
  __shared__ __align__(16) unsigned short Wt[32][40];       // 2.5 KB weight tile
  __shared__ float q2s[32];
  __shared__ float k2s[2][32];

  // krn = -d2 / (-2*gamma + 1e-6)  ==>  weight = exp(scale * d2)
  const float scale = -1.0f / (-2.0f * gamma[h] + 1e-6f);
  const size_t rowbase = (size_t)b * SLEN + (size_t)rt * 32;

  // issue async staging of K/V chunk `ck` into buffer `buf`
  auto stage_chunk = [&](int ck, int buf) {
    const uint64_t gk = (uint64_t)(uintptr_t)(Kh + ((size_t)b * SLEN + (size_t)ck * 32) * DDIM);
    const uint32_t lk = lds_addr32(&Ks[buf][0][0]);
#pragma unroll
    for (int it = 0; it < 8; ++it) {
      const int i = tid + it * 128;
      async_ld_b128(lk + i * 16, gk + (uint64_t)i * 16);
    }
    const uint64_t gv = (uint64_t)(uintptr_t)(Vt + (size_t)b * DDIM * SLEN + (size_t)ck * 32);
    const uint32_t lv = lds_addr32(&VsT[buf][0][0]);
#pragma unroll
    for (int it = 0; it < 8; ++it) {
      const int i = tid + it * 128;       // i in [0,1024): 4 b128 per d-row
      const int d = i >> 2, part = i & 3;
      async_ld_b128(lv + d * 80 + part * 16,
                    gv + ((uint64_t)d * SLEN + (uint64_t)part * 8) * 2);
    }
    if (tid < 32) k2s[buf][tid] = k2[(size_t)b * SLEN + (size_t)ck * 32 + tid];
  };

  // ---- prologue: stage Q tile + first K/V chunk ----
  {
    const uint64_t gq = (uint64_t)(uintptr_t)(Qh + rowbase * DDIM);
    const uint32_t lq = lds_addr32(&Qs[0][0]);
#pragma unroll
    for (int it = 0; it < 8; ++it) {
      const int i = tid + it * 128;
      async_ld_b128(lq + i * 16, gq + (uint64_t)i * 16);
    }
    if (tid < 32) q2s[tid] = q2[rowbase + tid];
    stage_chunk(0, 0);
    wait_async_all();
  }
  __syncthreads();

  // hoist Q A-fragments (invariant across key chunks): 8 x 8 VGPRs
  v16bf aq[8];
#pragma unroll
  for (int kk = 0; kk < 8; ++kk) aq[kk] = frag_a(&Qs[rhalf * 16][kk * 32], 256, lane);

  v8f zero = {};
  v8f acc[8];
#pragma unroll
  for (int t = 0; t < 8; ++t) acc[t] = zero;

  const int n  = lane & 15;
  const int hi = lane >> 4;

  for (int ck = 0; ck <= rt; ++ck) {
    const int cur = ck & 1;
    const bool pf = (ck < rt);
    if (pf) {                       // prefetch next chunk under this chunk's compute
      stage_chunk(ck + 1, cur ^ 1);
      wait_async_prev16();          // in-order completion: current chunk is resident
    } else {
      wait_async_all();
    }
    __syncthreads();

    // gram 16x16 quadrant, K-dim = 256; two interleaved K-fragment buffers
    v8f gacc = zero;
    {
      const unsigned short* kbase = &Ks[cur][chalf * 16][0];
      v16bf b0 = frag_b(kbase, 256, lane);
      v16bf b1 = frag_b(kbase + 32, 256, lane);
#pragma unroll
      for (int kk = 0; kk < 8; kk += 2) {
        gacc = __builtin_amdgcn_wmma_f32_16x16x32_bf16(
            false, aq[kk], false, b0, (short)0, gacc, false, false);
        if (kk + 2 < 8) b0 = frag_b(kbase + (kk + 2) * 32, 256, lane);
        gacc = __builtin_amdgcn_wmma_f32_16x16x32_bf16(
            false, aq[kk + 1], false, b1, (short)0, gacc, false, false);
        if (kk + 3 < 8) b1 = frag_b(kbase + (kk + 3) * 32, 256, lane);
      }
    }

    // d2 -> masked rbf weight -> LDS (bf16)
#pragma unroll
    for (int r = 0; r < 8; ++r) {
      const int m  = r + 8 * hi;
      const int sg = rt * 32 + rhalf * 16 + m;
      const int tg = ck * 32 + chalf * 16 + n;
      float d2 = q2s[rhalf * 16 + m] + k2s[cur][chalf * 16 + n] - 2.0f * gacc[r];
      d2 = fmaxf(d2, 0.0f);
      const float w = (tg <= sg) ? __expf(scale * d2) : 0.0f;
      Wt[rhalf * 16 + m][chalf * 16 + n] = f2bfbits(w);
    }
    __syncthreads();

    // out(16x128) += Wt(16x32) @ V(32x128); two interleaved V-fragment buffers
    {
      v16bf aw = frag_a(&Wt[rhalf * 16][0], 40, lane);
      v16bf b0 = frag_b(&VsT[cur][chalf * 128][0], 40, lane);
      v16bf b1 = frag_b(&VsT[cur][chalf * 128 + 16][0], 40, lane);
#pragma unroll
      for (int ct = 0; ct < 8; ct += 2) {
        acc[ct] = __builtin_amdgcn_wmma_f32_16x16x32_bf16(
            false, aw, false, b0, (short)0, acc[ct], false, false);
        if (ct + 2 < 8) b0 = frag_b(&VsT[cur][chalf * 128 + (ct + 2) * 16][0], 40, lane);
        acc[ct + 1] = __builtin_amdgcn_wmma_f32_16x16x32_bf16(
            false, aw, false, b1, (short)0, acc[ct + 1], false, false);
        if (ct + 3 < 8) b1 = frag_b(&VsT[cur][chalf * 128 + (ct + 3) * 16][0], 40, lane);
      }
    }
    __syncthreads();  // all readers of `cur` done before it is restaged next iter
  }

  // store f32 output [B,H,S,D]
#pragma unroll
  for (int r = 0; r < 8; ++r) {
    const int s = rt * 32 + rhalf * 16 + r + 8 * hi;
    float* o = out + (((size_t)b * NHEAD + h) * SLEN + s) * DDIM + chalf * 128 + n;
#pragma unroll
    for (int ct = 0; ct < 8; ++ct) o[ct * 16] = acc[ct][r];
  }
}

extern "C" void kernel_launch(void* const* d_in, const int* in_sizes, int n_in,
                              void* d_out, int out_size, void* d_ws, size_t ws_size,
                              hipStream_t stream) {
  // inputs: 0=x (unused), 1=e, 2=p, 3=W_q, 4=W_k, 5=W_v, 6=gamma
  const float* e  = (const float*)d_in[1];
  const float* p  = (const float*)d_in[2];
  const float* Wq = (const float*)d_in[3];
  const float* Wk = (const float*)d_in[4];
  const float* Wv = (const float*)d_in[5];
  const float* gamma = (const float*)d_in[6];
  float* out = (float*)d_out;

  char* ws = (char*)d_ws;
  const size_t matBytes = (size_t)BATCH * SLEN * DDIM * sizeof(unsigned short); // 2 MB
  unsigned short* Qh = (unsigned short*)(ws);
  unsigned short* Kh = (unsigned short*)(ws + matBytes);
  unsigned short* Vt = (unsigned short*)(ws + 2 * matBytes);   // [B][D][S]
  float* q2 = (float*)(ws + 3 * matBytes);
  float* k2 = (float*)(ws + 3 * matBytes + (size_t)BATCH * SLEN * sizeof(float));

  proj_kernel<<<dim3(BATCH * SLEN / 128, 3, 1), 256, 0, stream>>>(
      p, e, Wq, Wk, Wv, Qh, Kh, Vt, q2, k2);

  attn_kernel<<<dim3(SLEN / 32, NHEAD, BATCH), 128, 0, stream>>>(
      Qh, Kh, Vt, q2, k2, gamma, out);
}
